// GIKDWConv_61899068670114
// MI455X (gfx1250) — matrix operand order repaired and checked by reading, not compiled, not run on hardware
//
#include <hip/hip_runtime.h>

// CDNA5 / gfx1250: wave32, WMMA f32_16x16x32_f16
typedef _Float16 v16h __attribute__((ext_vector_type(16)));
typedef _Float16 h8   __attribute__((ext_vector_type(8)));
typedef _Float16 h4   __attribute__((ext_vector_type(4)));
typedef float    v8f  __attribute__((ext_vector_type(8)));

#define C_CH 384
#define N_B  16
#define H_IM 64
#define W_IM 64
#define LDSW 80   // 8 left pad + 64 data + 8 right pad (keeps b128 A-loads 16B aligned)
#define LDSH 70   // 3 top pad + 64 data + 3 bottom pad

__global__ __launch_bounds__(512)
void dwconv7_sym_wmma(const float* __restrict__ x,
                      const float* __restrict__ w,
                      float* __restrict__ out) {
    // input plane tile in f16 (halo zeros implement the conv zero padding)
    __shared__ __align__(16) _Float16 sx[LDSH * LDSW];          // 11200 B
    // symmetrized 7x7 weights (row stride 8)
    __shared__ __align__(16) _Float16 swh[7 * 8];               // 112 B
    // precomputed B fragments: [ky][khalf][n][e] -> banded matrix elements
    __shared__ __align__(16) _Float16 sbt[7 * 2 * 16 * 16];     // 7168 B

    const int c   = blockIdx.x;
    const int nb  = blockIdx.y;
    const int tid = threadIdx.x;

    const float* xp = x   + ((size_t)nb * C_CH + c) * (H_IM * W_IM);
    float*       op = out + ((size_t)nb * C_CH + c) * (H_IM * W_IM);

    // ---- Phase A: zero-fill LDS plane; build C4-symmetrized weights ----
    for (int i = tid; i < (LDSH * LDSW) / 8; i += 512) {
        ((uint4*)sx)[i] = make_uint4(0u, 0u, 0u, 0u);
    }
    if (tid < 49) {
        const int a = tid / 7, b = tid % 7;
        const float* wc = w + c * 49;
        // w_sym = mean of the 4 rot90 images of the 7x7 kernel
        float s = 0.25f * (wc[a * 7 + b] +
                           wc[b * 7 + (6 - a)] +
                           wc[(6 - a) * 7 + (6 - b)] +
                           wc[(6 - b) * 7 + a]);
        swh[a * 8 + b] = (_Float16)s;
    }
    __syncthreads();

    // ---- Phase B: f32 -> f16 plane into LDS interior; build B-fragment table ----
    // 4096 floats per plane, float4 loads (coalesced 16B), 2 iters/thread
    #pragma unroll
    for (int it = 0; it < 2; ++it) {
        const int e4 = tid + it * 512;               // float4 index, 0..1023
        const float4 v = ((const float4*)xp)[e4];
        const int flat = e4 * 4;
        const int row = flat >> 6;
        const int col = flat & 63;                   // multiple of 4 -> 8B-aligned LDS store
        h4 hv = { (_Float16)v.x, (_Float16)v.y, (_Float16)v.z, (_Float16)v.w };
        *(h4*)&sx[(row + 3) * LDSW + col + 8] = hv;
    }
    // B[kk, n] = w_sym[ky][kk - n - 5] for tap in [0,6]; contiguous k order used here: kk = khalf*16 + e
    for (int i = tid; i < 7 * 2 * 16 * 16; i += 512) {
        const int e  = i & 15;
        const int n  = (i >> 4) & 15;
        const int kh = (i >> 8) & 1;
        const int ky = i >> 9;
        const int tap = kh * 16 + e - n - 5;
        const int ct  = tap < 0 ? 0 : (tap > 6 ? 6 : tap);
        const _Float16 val = swh[ky * 8 + ct];
        sbt[i] = (tap >= 0 && tap <= 6) ? val : (_Float16)0.0f;
    }
    __syncthreads();

    // ---- Phase C: one 16x16 output tile per wave, 7 WMMAs (one per kernel row) ----
    const int lane = tid & 31;
    const int m    = lane & 15;   // A row / B col / C col index
    const int hf   = lane >> 4;   // K-half select per ISA fragment layouts
    const int wv   = tid >> 5;    // wave id 0..15
    const int y0   = (wv >> 2) * 16;
    const int w0   = (wv & 3) * 16;

    v8f acc = {};
    #pragma unroll
    for (int ky = 0; ky < 7; ++ky) {
        // A fragment: row (y0+m+ky-3) of the padded plane, cols w0-8 .. w0+23
        // LDS col index = input col + 8  ->  base half-index = y_row*80 + w0 + hf*16 (16B aligned)
        const _Float16* rp = &sx[(y0 + m + ky) * LDSW + w0 + hf * 16];
        h8 alo = *(const h8*)rp;            // K = hf*16 + 0..7     -> elements 0..7
        h8 ahi = *(const h8*)(rp + 8);      // K = hf*16 + 8..15    -> elements 8..15
        v16h a = __builtin_shufflevector(alo, ahi,
                 0,1,2,3,4,5,6,7,8,9,10,11,12,13,14,15);

        const _Float16* bp = &sbt[((ky * 2 + hf) * 16 + m) * 16];
        h8 blo = *(const h8*)bp;
        h8 bhi = *(const h8*)(bp + 8);
        v16h b = __builtin_shufflevector(blo, bhi,
                 0,1,2,3,4,5,6,7,8,9,10,11,12,13,14,15);

        acc = __builtin_amdgcn_wmma_f32_16x16x32_f16(
                  /*neg_a=*/false, a, /*neg_b=*/false, b,
                  /*c_mod=*/(short)0, acc,
                  /*reuse_a=*/false, /*reuse_b=*/false);
    }

    // C/D layout: element r -> row (r + 8*hf), column m; lanes 0..15 give 64B coalesced rows
    #pragma unroll
    for (int r = 0; r < 8; ++r) {
        op[(size_t)(y0 + r + hf * 8) * W_IM + (w0 + m)] = acc[r];
    }
}

extern "C" void kernel_launch(void* const* d_in, const int* in_sizes, int n_in,
                              void* d_out, int out_size, void* d_ws, size_t ws_size,
                              hipStream_t stream) {
    (void)in_sizes; (void)n_in; (void)d_ws; (void)ws_size; (void)out_size;
    const float* x = (const float*)d_in[0];   // [16, 384, 64, 64] fp32
    const float* w = (const float*)d_in[1];   // [384, 1, 7, 7]   fp32
    float* out = (float*)d_out;               // [16, 384, 64, 64] fp32

    dim3 grid(C_CH, N_B);   // one block per (n, c) plane
    hipLaunchKernelGGL(dwconv7_sym_wmma, grid, dim3(512), 0, stream, x, w, out);
}
